// CollaborativeAttention_11544872091773
// MI455X (gfx1250) — matrix operand order repaired and checked by reading, not compiled, not run on hardware
//
#include <hip/hip_runtime.h>

// ---------------------------------------------------------------------------
// CollaborativeAttention for MI455X (gfx1250, wave32, WMMA bf16 -> f32 acc)
// ---------------------------------------------------------------------------

#define BB 8
#define NN 1024
#define CC 768
#define HH 12
#define DH 64

typedef __bf16 bf16_t;
typedef __attribute__((ext_vector_type(16))) __bf16 v16bf;
typedef __attribute__((ext_vector_type(8)))  __bf16 v8bf;
typedef __attribute__((ext_vector_type(8)))  float  v8f;

// A-matrix fragment, 16x32 bf16 (MxK), rows contiguous in memory with stride ld.
// Lanes 0-15 hold M=lane, K={0..7,16..23}; lanes 16-31 hold K={8..15,24..31}.
__device__ __forceinline__ v16bf load_a_frag(const bf16_t* base, int ld, int lane) {
  const int r = lane & 15;
  const int g = (lane >> 4) & 1;
  const bf16_t* p = base + (size_t)r * ld + (g << 3);
  v8bf lo = *(const v8bf*)p;         // K = 8g .. 8g+7
  v8bf hi = *(const v8bf*)(p + 16);  // K = 16+8g .. 16+8g+7
  return __builtin_shufflevector(lo, hi, 0, 1, 2, 3, 4, 5, 6, 7, 8, 9, 10, 11, 12, 13, 14, 15);
}

// B-matrix fragment, 32x16 bf16 (KxN), stored row-major [N][K] (B^T rows
// contiguous). Lanes 0-15: N=lane, K={0..7,16..23}; lanes 16-31: N=lane-16, K={8..15,24..31}.
__device__ __forceinline__ v16bf load_b_frag(const bf16_t* base, int ld, int lane) {
  const int n = lane & 15;
  const int g = (lane >> 4) & 1;
  const bf16_t* p = base + (size_t)n * ld + (g << 3);
  v8bf lo = *(const v8bf*)p;
  v8bf hi = *(const v8bf*)(p + 16);
  return __builtin_shufflevector(lo, hi, 0, 1, 2, 3, 4, 5, 6, 7, 8, 9, 10, 11, 12, 13, 14, 15);
}

__device__ __forceinline__ v8f wmma_bf16(v16bf a, v16bf b, v8f c) {
  return __builtin_amdgcn_wmma_f32_16x16x32_bf16(
      /*neg_a=*/false, a, /*neg_b=*/false, b,
      /*c_mod=*/(short)0, c, /*reuse_a=*/false, /*reuse_b=*/false);
}

// ---------------------------------------------------------------------------
// fp32 -> bf16 convert
// ---------------------------------------------------------------------------
__global__ void cvt_kernel(const float* __restrict__ in, bf16_t* __restrict__ out, int n) {
  int i = blockIdx.x * blockDim.x + threadIdx.x;
  if (i < n) out[i] = (bf16_t)in[i];
}

// ---------------------------------------------------------------------------
// Content bias: cb[(b*H+h)*N+m] = sum_d x[b,m,d] * Wcb[h,d]
// ---------------------------------------------------------------------------
__global__ void cbias_kernel(const float* __restrict__ x, const float* __restrict__ Wcb,
                             float* __restrict__ cbout) {
  int tid = blockIdx.x * blockDim.x + threadIdx.x;
  if (tid >= BB * HH * NN) return;
  int m = tid & (NN - 1);
  int h = (tid >> 10) % HH;
  int b = tid / (NN * HH);
  const float* xr = x + ((size_t)b * NN + m) * CC;
  const float* w  = Wcb + (size_t)h * CC;
  float s = 0.f;
  for (int d = 0; d < CC; ++d) s += xr[d] * w[d];
  cbout[tid] = s;
}

// ---------------------------------------------------------------------------
// WMMA GEMM: C[M,Ncols] = A[M,K](bf16) x W[Ncols,K]^T(bf16)
// One wave computes a 32x64 output tile (B fragments reused across 2 row tiles).
//  mode 0: out bf16 row-major, no bias
//  mode 1: out = per-head-transposed V: bf16 Vt[b,h,dh,n], bias added
//  mode 2: out fp32 row-major, bias added (final projection)
// ---------------------------------------------------------------------------
__global__ __launch_bounds__(32) void gemm32x64(const bf16_t* __restrict__ A,
                                                const bf16_t* __restrict__ W,
                                                const float* __restrict__ bias,
                                                void* __restrict__ outp,
                                                int M, int Ncols, int K, int mode) {
  const int lane = threadIdx.x;
  const int m0 = blockIdx.x * 32;
  const int c0 = blockIdx.y * 64;
  v8f acc[2][4];
#pragma unroll
  for (int r = 0; r < 2; ++r)
#pragma unroll
    for (int t = 0; t < 4; ++t) acc[r][t] = (v8f){};
  const bf16_t* arow0 = A + (size_t)m0 * K;
  const bf16_t* arow1 = A + (size_t)(m0 + 16) * K;
  const bf16_t* w0    = W + (size_t)c0 * K;
  for (int k0 = 0; k0 < K; k0 += 32) {
    v16bf a0 = load_a_frag(arow0 + k0, K, lane);
    v16bf a1 = load_a_frag(arow1 + k0, K, lane);
#pragma unroll
    for (int t = 0; t < 4; ++t) {
      v16bf bw = load_b_frag(w0 + (size_t)(t * 16) * K + k0, K, lane);
      acc[0][t] = wmma_bf16(a0, bw, acc[0][t]);
      acc[1][t] = wmma_bf16(a1, bw, acc[1][t]);
    }
  }
  const int cl = lane & 15;
  const int rhalf = (lane >> 4) << 3;
#pragma unroll
  for (int r = 0; r < 2; ++r) {
#pragma unroll
    for (int t = 0; t < 4; ++t) {
#pragma unroll
      for (int i = 0; i < 8; ++i) {
        int row = m0 + 16 * r + rhalf + i;
        int col = c0 + t * 16 + cl;
        float v = acc[r][t][i];
        if (mode == 0) {
          ((bf16_t*)outp)[(size_t)row * Ncols + col] = (bf16_t)v;
        } else if (mode == 1) {
          v += bias[col];
          int bb = row >> 10, n = row & 1023;
          int hh = col >> 6,  dh = col & 63;
          ((bf16_t*)outp)[(((size_t)bb * HH + hh) * DH + dh) * NN + n] = (bf16_t)v;
        } else {
          v += bias[col];
          ((float*)outp)[(size_t)row * Ncols + col] = v;
        }
      }
    }
  }
}

// ---------------------------------------------------------------------------
// Async-stage one 64(dh) x 32(m) bf16 V tile (4 KB) into LDS.
// 256 16-byte chunks, 8 per lane, via global_load_async_to_lds_b128 (ASYNCcnt).
// ---------------------------------------------------------------------------
__device__ __forceinline__ void async_stage_vtile(const bf16_t* gbase, unsigned lds_base,
                                                  int lane) {
#pragma unroll
  for (int i = 0; i < 8; ++i) {
    const int c = lane + 32 * i;        // chunk 0..255
    const int row = c >> 2;             // dh row (64B each = 4 chunks)
    const int colb = (c & 3) << 4;      // byte offset within row
    const char* g = (const char*)gbase + (size_t)row * (NN * 2) + colb;
    const unsigned l = lds_base + ((unsigned)c << 4);
    asm volatile("global_load_async_to_lds_b128 %0, %1, off"
                 :: "v"(l), "v"(g) : "memory");
  }
}

// ---------------------------------------------------------------------------
// Flash attention: one wave handles 16 query rows of one (b,h).
// Q (Wmix-scaled, bf16) in LDS; K streamed from L2; V async-DMA'd to LDS
// (double-buffered, overlapped with the score WMMAs).
// ---------------------------------------------------------------------------
__global__ __launch_bounds__(32) void attn_kernel(const bf16_t* __restrict__ Qs,
                                                  const bf16_t* __restrict__ Ks,
                                                  const bf16_t* __restrict__ Vt,
                                                  const float* __restrict__ cb,
                                                  const float* __restrict__ Wmix,
                                                  bf16_t* __restrict__ AO) {
  __shared__ __attribute__((aligned(16))) bf16_t Qlds[16 * CC];
  __shared__ __attribute__((aligned(16))) bf16_t Plds[16 * 32];
  __shared__ __attribute__((aligned(16))) bf16_t Vlds[2][DH * 32];
  const int lane = threadIdx.x;
  const int bid = blockIdx.x;
  const int qt = bid & 63;          // N/16 tiles
  const int h  = (bid >> 6) % HH;
  const int b  = bid / (64 * HH);
  const int n0 = qt * 16;

  const bf16_t* kb  = Ks + (size_t)b * NN * CC;
  const bf16_t* vb  = Vt + ((size_t)b * HH + h) * (size_t)DH * NN;
  const float*  cbb = cb + ((size_t)b * HH + h) * NN;
  const float SCALE = 0.125f;  // HEAD_DIM^-0.5 = 64^-0.5
  const unsigned vlds0 = (unsigned)(uintptr_t)&Vlds[0][0];
  const unsigned vlds1 = (unsigned)(uintptr_t)&Vlds[1][0];

  // Kick off DMA of the first V tile before staging Q.
  async_stage_vtile(vb, vlds0, lane);

  // Stage per-head query tile: Qlds[r][d] = q_shared[b,n0+r,d] * Wmix[h,d]
  const bf16_t* qsrc = Qs + ((size_t)b * NN + n0) * CC;
  const float*  wm   = Wmix + (size_t)h * CC;
  for (int r = 0; r < 16; ++r) {
    for (int d = lane * 4; d < CC; d += 128) {
#pragma unroll
      for (int j = 0; j < 4; ++j) {
        float qv = (float)qsrc[(size_t)r * CC + d + j];
        Qlds[r * CC + d + j] = (bf16_t)(qv * wm[d + j]);
      }
    }
  }
  __syncthreads();

  float m_i[8], l_i[8];
#pragma unroll
  for (int i = 0; i < 8; ++i) { m_i[i] = -3.0e38f; l_i[i] = 0.f; }
  v8f o0 = {}, o1 = {}, o2 = {}, o3 = {};

  for (int m0 = 0; m0 < NN; m0 += 32) {
    const int buf = (m0 >> 5) & 1;
    // Prefetch-issue next V tile into the other buffer (overlaps S compute).
    if (m0 + 32 < NN)
      async_stage_vtile(vb + (size_t)(m0 + 32), buf ? vlds0 : vlds1, lane);
    if (m0 + 32 < NN) __builtin_prefetch(kb + (size_t)(m0 + 32) * CC, 0, 1);

    // ---- S = Qmix . K^T over full key dim (768) ----
    v8f s0 = {}, s1 = {};
    for (int d0 = 0; d0 < CC; d0 += 32) {
      v16bf a   = load_a_frag(Qlds + d0, CC, lane);
      v16bf k0f = load_b_frag(kb + (size_t)(m0 +  0) * CC + d0, CC, lane);
      v16bf k1f = load_b_frag(kb + (size_t)(m0 + 16) * CC + d0, CC, lane);
      s0 = wmma_bf16(a, k0f, s0);
      s1 = wmma_bf16(a, k1f, s1);
    }
    // ---- bias + scale (lane owns one key column per tile) ----
    const float cb0 = cbb[m0 + (lane & 15)];
    const float cb1 = cbb[m0 + 16 + (lane & 15)];
    float rm[8];
#pragma unroll
    for (int i = 0; i < 8; ++i) {
      s0[i] = (s0[i] + cb0) * SCALE;
      s1[i] = (s1[i] + cb1) * SCALE;
      rm[i] = fmaxf(s0[i], s1[i]);
    }
#pragma unroll
    for (int mask = 1; mask <= 8; mask <<= 1)
#pragma unroll
      for (int i = 0; i < 8; ++i)
        rm[i] = fmaxf(rm[i], __shfl_xor(rm[i], mask, 32));
    float corr[8], rs[8];
#pragma unroll
    for (int i = 0; i < 8; ++i) {
      float mn = fmaxf(m_i[i], rm[i]);
      corr[i] = __expf(m_i[i] - mn);
      m_i[i]  = mn;
      float p0 = __expf(s0[i] - mn);
      float p1 = __expf(s1[i] - mn);
      s0[i] = p0; s1[i] = p1;
      rs[i] = p0 + p1;
    }
#pragma unroll
    for (int mask = 1; mask <= 8; mask <<= 1)
#pragma unroll
      for (int i = 0; i < 8; ++i)
        rs[i] += __shfl_xor(rs[i], mask, 32);
#pragma unroll
    for (int i = 0; i < 8; ++i) {
      l_i[i] = l_i[i] * corr[i] + rs[i];
      o0[i] *= corr[i]; o1[i] *= corr[i]; o2[i] *= corr[i]; o3[i] *= corr[i];
    }
    // ---- relayout P: C/D layout -> row-major LDS -> A fragment ----
    const int prow = (lane >> 4) << 3;
    const int pcol = lane & 15;
#pragma unroll
    for (int i = 0; i < 8; ++i) {
      Plds[(prow + i) * 32 + pcol]      = (bf16_t)s0[i];
      Plds[(prow + i) * 32 + 16 + pcol] = (bf16_t)s1[i];
    }
    __syncthreads();
    // Current V tile must have landed: with <=8 (next tile) outstanding, the
    // in-order async queue guarantees this tile's 8 chunks completed.
    if (m0 + 32 < NN)
      asm volatile("s_wait_asynccnt 0x8" ::: "memory");
    else
      asm volatile("s_wait_asynccnt 0x0" ::: "memory");
    const bf16_t* vtile = &Vlds[buf][0];
    v16bf ap = load_a_frag(Plds, 32, lane);
    o0 = wmma_bf16(ap, load_b_frag(vtile + 0 * 16 * 32, 32, lane), o0);
    o1 = wmma_bf16(ap, load_b_frag(vtile + 1 * 16 * 32, 32, lane), o1);
    o2 = wmma_bf16(ap, load_b_frag(vtile + 2 * 16 * 32, 32, lane), o2);
    o3 = wmma_bf16(ap, load_b_frag(vtile + 3 * 16 * 32, 32, lane), o3);
    __syncthreads();
  }

  // ---- normalize + store bf16 for the projection GEMM ----
  const int rhalf = (lane >> 4) << 3;
  const int cl = lane & 15;
  v8f oo[4] = {o0, o1, o2, o3};
#pragma unroll
  for (int t = 0; t < 4; ++t)
#pragma unroll
    for (int i = 0; i < 8; ++i) {
      int n = n0 + rhalf + i;
      int c = h * DH + t * 16 + cl;
      AO[((size_t)b * NN + n) * CC + c] = (bf16_t)(oo[t][i] / l_i[i]);
    }
}

// ---------------------------------------------------------------------------
// Host launch
// ---------------------------------------------------------------------------
extern "C" void kernel_launch(void* const* d_in, const int* in_sizes, int n_in,
                              void* d_out, int out_size, void* d_ws, size_t ws_size,
                              hipStream_t stream) {
  (void)in_sizes; (void)n_in; (void)out_size; (void)ws_size;
  const float* x     = (const float*)d_in[0];
  const float* Wq    = (const float*)d_in[1];
  const float* Wk    = (const float*)d_in[2];
  const float* Wv    = (const float*)d_in[3];
  const float* bv    = (const float*)d_in[4];
  const float* Wmix  = (const float*)d_in[5];
  const float* Wcb   = (const float*)d_in[6];
  const float* Wproj = (const float*)d_in[7];
  const float* bproj = (const float*)d_in[8];

  char* ws = (char*)d_ws;
  size_t off = 0;
  auto carve = [&](size_t bytes) -> void* {
    void* p = ws + off;
    off += (bytes + 255) & ~(size_t)255;
    return p;
  };
  const size_t Mtot = (size_t)BB * NN;  // 8192
  bf16_t* xbf  = (bf16_t*)carve(Mtot * CC * 2);
  bf16_t* wqb  = (bf16_t*)carve((size_t)CC * CC * 2);
  bf16_t* wkb  = (bf16_t*)carve((size_t)CC * CC * 2);
  bf16_t* wvb  = (bf16_t*)carve((size_t)CC * CC * 2);
  bf16_t* wpb  = (bf16_t*)carve((size_t)CC * CC * 2);
  bf16_t* Qsb  = (bf16_t*)carve(Mtot * CC * 2);
  bf16_t* Ksb  = (bf16_t*)carve(Mtot * CC * 2);
  bf16_t* Vtb  = (bf16_t*)carve(Mtot * CC * 2);
  bf16_t* AO   = (bf16_t*)carve(Mtot * CC * 2);
  float*  cbuf = (float*)carve((size_t)BB * HH * NN * 4);

  const int nX = (int)(Mtot * CC);
  const int nW = CC * CC;
  cvt_kernel<<<(nX + 255) / 256, 256, 0, stream>>>(x, xbf, nX);
  cvt_kernel<<<(nW + 255) / 256, 256, 0, stream>>>(Wq, wqb, nW);
  cvt_kernel<<<(nW + 255) / 256, 256, 0, stream>>>(Wk, wkb, nW);
  cvt_kernel<<<(nW + 255) / 256, 256, 0, stream>>>(Wv, wvb, nW);
  cvt_kernel<<<(nW + 255) / 256, 256, 0, stream>>>(Wproj, wpb, nW);

  dim3 ggrid((unsigned)(Mtot / 32), (unsigned)(CC / 64));  // 256 x 12
  gemm32x64<<<ggrid, 32, 0, stream>>>(xbf, wqb, nullptr, Qsb, (int)Mtot, CC, CC, 0);
  gemm32x64<<<ggrid, 32, 0, stream>>>(xbf, wkb, nullptr, Ksb, (int)Mtot, CC, CC, 0);
  gemm32x64<<<ggrid, 32, 0, stream>>>(xbf, wvb, bv, Vtb, (int)Mtot, CC, CC, 1);

  cbias_kernel<<<(BB * HH * NN + 255) / 256, 256, 0, stream>>>(x, Wcb, cbuf);

  attn_kernel<<<BB * HH * (NN / 16), 32, 0, stream>>>(Qsb, Ksb, Vtb, cbuf, Wmix, AO);

  gemm32x64<<<ggrid, 32, 0, stream>>>(AO, wpb, bproj, d_out, (int)Mtot, CC, CC, 2);
}
